// FlexibleGNN_10299331576465
// MI455X (gfx1250) — matrix operand order repaired and checked by reading, not compiled, not run on hardware
//
#include <hip/hip_runtime.h>
#include <hip/hip_bf16.h>

// ---------------- problem constants (from reference) ----------------
#define NODES   50000
#define EDGES   800000
#define NODE_IN 128
#define EDGE_IN 16
#define HDIM    64
#define LAYERS  4
#define CLASSES 10
#define GRAPHS  128

typedef __attribute__((ext_vector_type(16))) __bf16 v16bf;
typedef __attribute__((ext_vector_type(8)))  float  v8f;

// fp32 -> bf16; native fptrunc lets the backend use packed v_cvt on gfx1250
__device__ __forceinline__ __bf16 f2bf(float f) {
    return (__bf16)f;
}

__device__ __forceinline__ v8f v8f_zero() {
    v8f z;
#pragma unroll
    for (int i = 0; i < 8; ++i) z[i] = 0.0f;
    return z;
}

__device__ __forceinline__ v8f wmma_bf16(v16bf a, v16bf b, v8f c) {
    // D = A(16x32 bf16) x B(32x16 bf16) + C(16x16 f32)
    return __builtin_amdgcn_wmma_f32_16x16x32_bf16(
        /*neg_a=*/false, a, /*neg_b=*/false, b,
        /*c_mod=*/(short)0, c, /*reuse_a=*/false, /*reuse_b=*/false);
}

// ---------------------------------------------------------------
// Weight shuffle: fp32 [K x 64] row-major -> bf16 WMMA B-fragments.
// Fragment order: [s (K/32)][t (N/16)][lane 0..31][i 0..15] so each lane
// loads its 16 bf16 (32B) contiguously.
// Dense 16-bit B (32x16) layout per ISA B-matrix tables:
//   lanes 0-15:  col = lane,    K = 0..7, 16..23  (2 per VGPR)
//   lanes 16-31: col = lane-16, K = 8..15, 24..31
// ---------------------------------------------------------------
__device__ __forceinline__ void frag_store(const float* __restrict__ W,
                                           __bf16* __restrict__ dst, int f) {
    int i    = f & 15;
    int lane = (f >> 4) & 31;
    int t    = (f >> 9) & 3;
    int s    = f >> 11;
    int k = 32 * s + ((lane >> 4) & 1) * 8 + i + ((i >> 3) << 3);
    int n = 16 * t + (lane & 15);
    dst[f] = f2bf(W[k * HDIM + n]);
}

__global__ void __launch_bounds__(256)
convert_weights(const float* __restrict__ w_node,
                const float* __restrict__ w1,
                const float* __restrict__ w2,
                __bf16* __restrict__ wnf,
                __bf16* __restrict__ w1f,
                __bf16* __restrict__ w2f) {
    int idx = blockIdx.x * 256 + threadIdx.x;
    if (idx < 8192) {                      // w_node: K=128 -> 4 k-blocks
        frag_store(w_node, wnf, idx);
    } else if (idx < 8192 + 16384) {       // w1: 4 layers x (2 k-blocks)
        int j = idx - 8192;
        int layer = j >> 12;
        frag_store(w1 + layer * (HDIM * HDIM), w1f + layer * 4096, j & 4095);
    } else if (idx < 8192 + 32768) {       // w2
        int j = idx - 8192 - 16384;
        int layer = j >> 12;
        frag_store(w2 + layer * (HDIM * HDIM), w2f + layer * 4096, j & 4095);
    }
}

// ---------------------------------------------------------------
// Node embedding: h[N,64] = x[N,128] @ w_node + b_node
// One wave -> 16 rows x 64 cols; 4 K-steps x 4 N-tiles = 16 WMMAs.
// A-frag (16-bit A 16x32, ISA 7.12.2): lane<16 -> row=lane, K={0..7,16..23};
// lane>=16 -> row=lane-16, K={8..15,24..31}.
// ---------------------------------------------------------------
__global__ void __launch_bounds__(128)
embed_kernel(const float* __restrict__ x, const __bf16* __restrict__ wnf,
             const float* __restrict__ b_node, float* __restrict__ h) {
    const int lane = threadIdx.x & 31;
    const int wave = blockIdx.x * 4 + (threadIdx.x >> 5);
    const int row0 = wave * 16;
    if (row0 >= NODES) return;

    const int row = row0 + (lane & 15);
    const int kb8 = (lane & 16) ? 8 : 0;

    v8f acc[4];
#pragma unroll
    for (int t = 0; t < 4; ++t) acc[t] = v8f_zero();

#pragma unroll
    for (int s = 0; s < 4; ++s) {
        const float* p = x + row * NODE_IN + s * 32 + kb8;
        v16bf a;
#pragma unroll
        for (int i = 0; i < 8; ++i) a[i] = f2bf(p[i]);
#pragma unroll
        for (int i = 0; i < 8; ++i) a[8 + i] = f2bf(p[16 + i]);
#pragma unroll
        for (int t = 0; t < 4; ++t) {
            v16bf b = *(const v16bf*)(wnf + (((s * 4 + t) * 32 + lane) << 4));
            acc[t] = wmma_bf16(a, b, acc[t]);
        }
    }

    const int mrow0 = row0 + ((lane & 16) ? 8 : 0);
    const int col = lane & 15;
#pragma unroll
    for (int t = 0; t < 4; ++t) {
        int n = t * 16 + col;
        float bn = b_node[n];
#pragma unroll
        for (int r = 0; r < 8; ++r)
            h[(mrow0 + r) * HDIM + n] = acc[t][r] + bn;
    }
}

// ---------------------------------------------------------------
// Fused edge pass: agg[dst] += relu(h[src] + edge_attr @ w_edge + b_edge)
// One wave per edge; lane handles 2 channels; edge_attr broadcast by shfl.
// ---------------------------------------------------------------
__global__ void __launch_bounds__(256)
edge_kernel(const float* __restrict__ h, const float* __restrict__ edge_attr,
            const int* __restrict__ srcI, const int* __restrict__ dstI,
            const float* __restrict__ w_edge, const float* __restrict__ b_edge,
            float* __restrict__ agg) {
    __shared__ float wsh[EDGE_IN * HDIM];
    __shared__ float bsh[HDIM];
    for (int i = threadIdx.x; i < EDGE_IN * HDIM; i += 256) wsh[i] = w_edge[i];
    if (threadIdx.x < HDIM) bsh[threadIdx.x] = b_edge[threadIdx.x];
    __syncthreads();

    const int lane   = threadIdx.x & 31;
    const int wave   = blockIdx.x * 8 + (threadIdx.x >> 5);
    const int nwaves = gridDim.x * 8;
    const int c0 = lane * 2;
    const float bias0 = bsh[c0], bias1 = bsh[c0 + 1];

    for (int e = wave; e < EDGES; e += nwaves) {
        float eav = (lane < EDGE_IN) ? edge_attr[(size_t)e * EDGE_IN + lane] : 0.0f;
        if (e + nwaves < EDGES)
            __builtin_prefetch(edge_attr + (size_t)(e + nwaves) * EDGE_IN, 0, 1);

        int sn = srcI[e];
        int dn = dstI[e];
        float2 hv = *(const float2*)(h + (size_t)sn * HDIM + c0);
        float m0 = bias0 + hv.x;
        float m1 = bias1 + hv.y;
#pragma unroll
        for (int k = 0; k < EDGE_IN; ++k) {
            float ek = __shfl(eav, k, 32);
            m0 = fmaf(ek, wsh[k * HDIM + c0],     m0);
            m1 = fmaf(ek, wsh[k * HDIM + c0 + 1], m1);
        }
        m0 = fmaxf(m0, 0.0f);
        m1 = fmaxf(m1, 0.0f);
        atomicAdd(&agg[(size_t)dn * HDIM + c0],     m0);
        atomicAdd(&agg[(size_t)dn * HDIM + c0 + 1], m1);
    }
}

// ---------------------------------------------------------------
// Fused GIN MLP: h = relu( relu((agg+h) @ W1 + b1) @ W2 + b2 )
// One wave -> 16 rows; LDS round-trip converts C/D layout -> A layout.
// ---------------------------------------------------------------
__global__ void __launch_bounds__(128)
mlp_kernel(const float* __restrict__ agg, float* __restrict__ h,
           const __bf16* __restrict__ w1f, const float* __restrict__ b1,
           const __bf16* __restrict__ w2f, const float* __restrict__ b2) {
    __shared__ float zs[4][16 * HDIM];
    const int lane  = threadIdx.x & 31;
    const int wslot = threadIdx.x >> 5;
    const int wave  = blockIdx.x * 4 + wslot;
    const int row0  = wave * 16;
    const bool active = (row0 < NODES);        // wave-uniform
    const int kb8   = (lane & 16) ? 8 : 0;
    const int col   = lane & 15;
    const int mbase = (lane & 16) ? 8 : 0;

    if (active) {
        const int row = row0 + col;
        v8f acc[4];
#pragma unroll
        for (int t = 0; t < 4; ++t) acc[t] = v8f_zero();
#pragma unroll
        for (int s = 0; s < 2; ++s) {
            const float* pa = agg + (size_t)row * HDIM + s * 32 + kb8;
            const float* ph = h   + (size_t)row * HDIM + s * 32 + kb8;
            v16bf a;
#pragma unroll
            for (int i = 0; i < 8; ++i) a[i]     = f2bf(pa[i] + ph[i]);
#pragma unroll
            for (int i = 0; i < 8; ++i) a[8 + i] = f2bf(pa[16 + i] + ph[16 + i]);
#pragma unroll
            for (int t = 0; t < 4; ++t) {
                v16bf b = *(const v16bf*)(w1f + (((s * 4 + t) * 32 + lane) << 4));
                acc[t] = wmma_bf16(a, b, acc[t]);
            }
        }
        // bias + relu, park z1 in LDS in row-major order
#pragma unroll
        for (int t = 0; t < 4; ++t) {
            int n = t * 16 + col;
            float bb = b1[n];
#pragma unroll
            for (int r = 0; r < 8; ++r)
                zs[wslot][(mbase + r) * HDIM + n] = fmaxf(acc[t][r] + bb, 0.0f);
        }
    }
    __syncthreads();
    if (active) {
        v8f acc[4];
#pragma unroll
        for (int t = 0; t < 4; ++t) acc[t] = v8f_zero();
#pragma unroll
        for (int s = 0; s < 2; ++s) {
            const float* p = &zs[wslot][col * HDIM + s * 32 + kb8];
            v16bf a;
#pragma unroll
            for (int i = 0; i < 8; ++i) a[i]     = f2bf(p[i]);
#pragma unroll
            for (int i = 0; i < 8; ++i) a[8 + i] = f2bf(p[16 + i]);
#pragma unroll
            for (int t = 0; t < 4; ++t) {
                v16bf b = *(const v16bf*)(w2f + (((s * 4 + t) * 32 + lane) << 4));
                acc[t] = wmma_bf16(a, b, acc[t]);
            }
        }
#pragma unroll
        for (int t = 0; t < 4; ++t) {
            int n = t * 16 + col;
            float bb = b2[n];
#pragma unroll
            for (int r = 0; r < 8; ++r)
                h[(size_t)(row0 + mbase + r) * HDIM + n] = fmaxf(acc[t][r] + bb, 0.0f);
        }
    }
}

// ---------------------------------------------------------------
// global_mean_pool accumulation
// ---------------------------------------------------------------
__global__ void __launch_bounds__(256)
pool_kernel(const float* __restrict__ h, const int* __restrict__ batch,
            float* __restrict__ sums, float* __restrict__ cnt) {
    int idx = blockIdx.x * 256 + threadIdx.x;
    if (idx >= NODES * 32) return;
    int node = idx >> 5;
    int c0 = (idx & 31) * 2;
    int g = batch[node];
    float2 hv = *(const float2*)(h + (size_t)node * HDIM + c0);
    atomicAdd(&sums[g * HDIM + c0],     hv.x);
    atomicAdd(&sums[g * HDIM + c0 + 1], hv.y);
    if ((idx & 31) == 0) atomicAdd(&cnt[g], 1.0f);
}

__global__ void __launch_bounds__(256)
out_kernel(const float* __restrict__ sums, const float* __restrict__ cnt,
           const float* __restrict__ w_out, const float* __restrict__ b_out,
           float* __restrict__ out) {
    int idx = blockIdx.x * 256 + threadIdx.x;
    if (idx >= GRAPHS * CLASSES) return;
    int g = idx / CLASSES, c = idx % CLASSES;
    float inv = 1.0f / fmaxf(cnt[g], 1.0f);
    float acc = b_out[c];
#pragma unroll 8
    for (int k = 0; k < HDIM; ++k)
        acc = fmaf(sums[g * HDIM + k] * inv, w_out[k * CLASSES + c], acc);
    out[idx] = acc;
}

// ---------------------------------------------------------------
extern "C" void kernel_launch(void* const* d_in, const int* in_sizes, int n_in,
                              void* d_out, int out_size, void* d_ws, size_t ws_size,
                              hipStream_t stream) {
    const float* x         = (const float*)d_in[0];
    const float* edge_attr = (const float*)d_in[1];
    const int*   edge_idx  = (const int*)d_in[2];
    const int*   batch     = (const int*)d_in[3];
    const float* w_node    = (const float*)d_in[4];
    const float* b_node    = (const float*)d_in[5];
    const float* w_edge    = (const float*)d_in[6];
    const float* b_edge    = (const float*)d_in[7];
    const float* w1        = (const float*)d_in[8];
    const float* b1        = (const float*)d_in[9];
    const float* w2        = (const float*)d_in[10];
    const float* b2        = (const float*)d_in[11];
    const float* w_out     = (const float*)d_in[12];
    const float* b_out     = (const float*)d_in[13];
    float* out = (float*)d_out;

    // workspace layout (all offsets 256B aligned)
    char* ws = (char*)d_ws;
    float*  h    = (float*)(ws);                               // 12.8 MB
    float*  agg  = (float*)(ws + 12800000);                    // 12.8 MB
    float*  sums = (float*)(ws + 25600000);                    // 32 KB
    float*  cnt  = (float*)(ws + 25632768);                    // 512 B
    __bf16* wnf  = (__bf16*)(ws + 25633280);                   // 16 KB
    __bf16* w1f  = (__bf16*)(ws + 25633280 + 16384);           // 32 KB
    __bf16* w2f  = (__bf16*)(ws + 25633280 + 16384 + 32768);   // 32 KB

    const int* srcI = edge_idx;          // edge_index[0]
    const int* dstI = edge_idx + EDGES;  // edge_index[1]

    const int nWaveTiles = (NODES + 15) / 16;                  // 3125
    const int gemmBlocks = (nWaveTiles + 3) / 4;               // 4 waves/block

    convert_weights<<<160, 256, 0, stream>>>(w_node, w1, w2, wnf, w1f, w2f);
    embed_kernel<<<gemmBlocks, 128, 0, stream>>>(x, wnf, b_node, h);

    for (int l = 0; l < LAYERS; ++l) {
        hipMemsetAsync(agg, 0, (size_t)NODES * HDIM * sizeof(float), stream);
        edge_kernel<<<2048, 256, 0, stream>>>(h, edge_attr, srcI, dstI,
                                              w_edge, b_edge, agg);
        mlp_kernel<<<gemmBlocks, 128, 0, stream>>>(agg, h,
                                                   w1f + l * 4096, b1 + l * HDIM,
                                                   w2f + l * 4096, b2 + l * HDIM);
    }

    hipMemsetAsync(sums, 0, (size_t)(GRAPHS * HDIM + 128) * sizeof(float), stream);
    pool_kernel<<<(NODES * 32 + 255) / 256, 256, 0, stream>>>(h, batch, sums, cnt);
    out_kernel<<<(GRAPHS * CLASSES + 255) / 256, 256, 0, stream>>>(sums, cnt,
                                                                   w_out, b_out, out);
    (void)in_sizes; (void)n_in; (void)out_size; (void)ws_size;
}
